// Seq2Seq_46712064311345
// MI455X (gfx1250) — matrix operand order repaired and checked by reading, not compiled, not run on hardware
//
#include <hip/hip_runtime.h>

// ---------------------------------------------------------------------------
// Seq2Seq LSTM (B=512, T=96, F=32, H=512, P=48) for gfx1250 / MI455X.
// 32 persistent workgroups, each owning 16 batch rows for the whole sequence.
// Per step: gates = [x|h](16x544,bf16) x W(544x2048,bf16) via
// v_wmma_f32_16x16x32_bf16, fp32 cell state in VGPRs, double-buffered LDS A.
// Weight fragments packed [nt][gate][kt]; the K loop is deliberately rolled
// (#pragma unroll 1) to bound register pressure -- full unroll caused the
// scheduler to hoist all fragment loads and spill them to scratch.
// ---------------------------------------------------------------------------

typedef __attribute__((ext_vector_type(16))) __bf16        v16bf;
typedef __attribute__((ext_vector_type(8)))  float         v8f;
typedef __attribute__((ext_vector_type(4)))  unsigned int  u32x4;

union ABfrag { v16bf bf; u32x4 q[2]; };

__device__ __forceinline__ unsigned short f2bf(float f) {
  union { float f; unsigned int u; } v; v.f = f;
  unsigned int u = v.u + 0x7fffu + ((v.u >> 16) & 1u);   // round-to-nearest-even
  return (unsigned short)(u >> 16);
}

// sigmoid(x) = rcp(1 + exp(-x)) : v_mul + v_exp + v_add + v_rcp
__device__ __forceinline__ float fast_sig(float x) {
  return __builtin_amdgcn_rcpf(1.0f + __expf(-x));
}
// tanh(x) = 2*sigmoid(2x) - 1
__device__ __forceinline__ float fast_tanh_(float x) {
  return 2.0f * fast_sig(2.0f * x) - 1.0f;
}

constexpr int Bc = 512, Tc = 96, Fc = 32, Hc = 512, Pc = 48;
constexpr int KTOT = Fc + Hc;        // 544
constexpr int NKT  = KTOT / 32;      // 17 k-tiles for the gate GEMM
constexpr int ROWS = 552;            // padded LDS row stride in bf16 (1104 B, 16B aligned)

// ---------------------------------------------------------------------------
// Weight packer -> per-lane WMMA B fragments, layout [nt][gate][kt][lane][8dw].
// Fragment (nt,g,kt): covers weight rows n = g*(ntile*16) + nt*16 + 0..15 and
// K columns kt*32..kt*32+31 of [Wih | Whh]. Within a fragment: lanes 0-15 hold
// K+0..15 (2 bf16/dword, low K first) for column n = nt*16+(lane&15);
// lanes 16-31 hold K+16..31.
// ---------------------------------------------------------------------------
__global__ void pack_wmma_b(const float* __restrict__ Wih,
                            const float* __restrict__ Whh,
                            int Fdim, int Kh, int ngates, int ntile, int ktiles,
                            unsigned int* __restrict__ dst) {
  int gid = blockIdx.x * blockDim.x + threadIdx.x;
  int total = ntile * ngates * ktiles * 32;
  if (gid >= total) return;
  int lane = gid & 31;
  int rest = gid >> 5;
  int kt = rest % ktiles;  rest /= ktiles;
  int g  = rest % ngates;  rest /= ngates;
  int nt = rest;
  int n  = g * (ntile * 16) + nt * 16 + (lane & 15);
  int kb = kt * 32 + ((lane < 16) ? 0 : 16);
  int fi = (nt * ngates + g) * ktiles + kt;
  __attribute__((aligned(16))) unsigned int o[8];
#pragma unroll
  for (int d = 0; d < 8; ++d) {
    int k0 = kb + 2 * d;
    int k1 = k0 + 1;
    float v0 = (k0 < Fdim) ? Wih[n * Fdim + k0] : Whh[n * Kh + (k0 - Fdim)];
    float v1 = (k1 < Fdim) ? Wih[n * Fdim + k1] : Whh[n * Kh + (k1 - Fdim)];
    o[d] = (unsigned int)f2bf(v0) | ((unsigned int)f2bf(v1) << 16);
  }
  unsigned int* p = dst + ((size_t)fi * 256 + (size_t)lane * 8);
  const u32x4* q = (const u32x4*)o;
  *(u32x4*)(p)     = q[0];
  *(u32x4*)(p + 4) = q[1];
}

// ---------------------------------------------------------------------------
// One LSTM step for one wave: gates for its 64 hidden units (4 N-tiles x
// 4 gates), fp32 cell state updated in VGPRs, bf16 h written to next A buffer.
// Rolled K loop: one advancing base pointer, gates at fixed immediates.
// ---------------------------------------------------------------------------
__device__ __forceinline__ void lstm_wave_step(
    const u32x4* __restrict__ Wq, const float* __restrict__ bias,
    const unsigned short* __restrict__ Ar, unsigned short* __restrict__ Aw,
    v8f cst[4], int wave, int lane) {
  const int n    = lane & 15;
  const int msel = (lane < 16) ? 0 : 8;   // C/D layout: M = vgpr + msel
  const int aoff = (lane < 16) ? 0 : 16;  // A layout half-wave K offset (shorts)
  const unsigned short* arow = Ar + n * ROWS + aoff;   // invariant A row base

#pragma unroll
  for (int tt = 0; tt < 4; ++tt) {
    const int jb  = wave * 64 + tt * 16;  // hidden-unit base for this tile
    const int ntb = wave * 4 + tt;        // N-tile index within one gate block
    // Base of this tile's [gate][kt] fragment block (u32x4 units).
    const u32x4* wp = Wq + (size_t)(ntb * 4 * NKT) * 64 + (size_t)lane * 2;
    const unsigned short* ap = arow;
    v8f acc0 = {}, acc1 = {}, acc2 = {}, acc3 = {};
#pragma unroll 1
    for (int kt = 0; kt < NKT; ++kt) {
      ABfrag a;
      a.q[0] = *(const u32x4*)(ap);        // ds offset 0
      a.q[1] = *(const u32x4*)(ap + 8);    // ds offset 16
      ABfrag b0, b1, b2, b3;
      b0.q[0] = wp[0 * NKT * 64 + 0];
      b0.q[1] = wp[0 * NKT * 64 + 1];
      b1.q[0] = wp[1 * NKT * 64 + 0];
      b1.q[1] = wp[1 * NKT * 64 + 1];
      b2.q[0] = wp[2 * NKT * 64 + 0];
      b2.q[1] = wp[2 * NKT * 64 + 1];
      b3.q[0] = wp[3 * NKT * 64 + 0];
      b3.q[1] = wp[3 * NKT * 64 + 1];
      acc0 = __builtin_amdgcn_wmma_f32_16x16x32_bf16(false, a.bf, false, b0.bf,
                                                     (short)0, acc0, false, false);
      acc1 = __builtin_amdgcn_wmma_f32_16x16x32_bf16(false, a.bf, false, b1.bf,
                                                     (short)0, acc1, false, false);
      acc2 = __builtin_amdgcn_wmma_f32_16x16x32_bf16(false, a.bf, false, b2.bf,
                                                     (short)0, acc2, false, false);
      acc3 = __builtin_amdgcn_wmma_f32_16x16x32_bf16(false, a.bf, false, b3.bf,
                                                     (short)0, acc3, false, false);
      wp += 64;        // next k-tile fragment (1024 B)
      ap += 32;        // next 32 bf16 of K in LDS (64 B)
    }
    const float bi = bias[       jb + n];
    const float bf = bias[ 512 + jb + n];
    const float bg = bias[1024 + jb + n];
    const float bo = bias[1536 + jb + n];
    v8f c = cst[tt];
#pragma unroll
    for (int e = 0; e < 8; ++e) {
      float iv = fast_sig(acc0[e] + bi);
      float fv = fast_sig(acc1[e] + bf);
      float gv = fast_tanh_(acc2[e] + bg);
      float ov = fast_sig(acc3[e] + bo);
      float cv = fv * c[e] + iv * gv;
      c[e] = cv;
      float hv = ov * fast_tanh_(cv);
      Aw[(e + msel) * ROWS + Fc + jb + n] = f2bf(hv);   // bf16 h for next step
    }
    cst[tt] = c;
  }
}

// ---------------------------------------------------------------------------
// Decoder output projection: out = h @ lin_W.T + lin_b (N=32 -> waves 0,1).
// Reads bf16 h from A, writes fp32 d_out and bf16 feedback x into A cols 0..31.
// ---------------------------------------------------------------------------
__device__ __forceinline__ void proj_step(
    const u32x4* __restrict__ Lq, const float* __restrict__ lin_b,
    unsigned short* __restrict__ A, float* __restrict__ out,
    int p, int m0, int wave, int lane) {
  const int n    = lane & 15;
  const int msel = (lane < 16) ? 0 : 8;
  const int aoff = (lane < 16) ? 0 : 16;
  const unsigned short* ap = A + n * ROWS + Fc + aoff;
  const u32x4* wp = Lq + (size_t)(wave * 16) * 64 + (size_t)lane * 2;
  v8f acc = {};
#pragma unroll 1
  for (int kk = 0; kk < 16; ++kk) {
    ABfrag a;
    a.q[0] = *(const u32x4*)(ap);
    a.q[1] = *(const u32x4*)(ap + 8);
    ABfrag b;
    b.q[0] = wp[0];
    b.q[1] = wp[1];
    acc = __builtin_amdgcn_wmma_f32_16x16x32_bf16(false, a.bf, false, b.bf,
                                                  (short)0, acc, false, false);
    wp += 64;
    ap += 32;
  }
  const int ncol = wave * 16 + n;
  const float bb = lin_b[ncol];
#pragma unroll
  for (int e = 0; e < 8; ++e) {
    const int m = e + msel;
    float v = acc[e] + bb;
    out[(size_t)(m0 + m) * (Pc * Fc) + (size_t)p * Fc + ncol] = v;
    A[m * ROWS + ncol] = f2bf(v);                    // feedback x for next step
  }
}

// ---------------------------------------------------------------------------
// Persistent per-batch-slice kernel: full encoder + decoder sequence.
// ---------------------------------------------------------------------------
__global__ __launch_bounds__(256)
void seq2seq_kernel(const float* __restrict__ input,
                    const float* __restrict__ enc_b,
                    const float* __restrict__ dec_b,
                    const float* __restrict__ lin_b,
                    const u32x4* __restrict__ encW,
                    const u32x4* __restrict__ decW,
                    const u32x4* __restrict__ linW,
                    float* __restrict__ out) {
  __shared__ __align__(16) unsigned short Abuf[2][16 * ROWS];
  const int tid  = threadIdx.x;
  const int wave = tid >> 5;
  const int lane = tid & 31;
  const int m0   = blockIdx.x * 16;

  // Init buffer 0: h = 0, x = input[:, 0, :]
  for (int idx = tid; idx < 16 * Hc; idx += 256) {
    Abuf[0][(idx >> 9) * ROWS + Fc + (idx & 511)] = 0;
  }
  for (int idx = tid; idx < 16 * Fc; idx += 256) {
    const int r = idx >> 5, c = idx & 31;
    Abuf[0][r * ROWS + c] =
        f2bf(input[(size_t)(m0 + r) * (Tc * Fc) + c]);
  }
  __syncthreads();

  v8f zero = {};
  v8f cst[4];
  cst[0] = zero; cst[1] = zero; cst[2] = zero; cst[3] = zero;

  // -------- Encoder: 96 steps, 1 barrier/step (double-buffered A) --------
  for (int t = 0; t < Tc; ++t) {
    const unsigned short* Ac = Abuf[t & 1];
    unsigned short*       An = Abuf[(t + 1) & 1];
    // Stage next x (t=95 stages input[:,95,:] as the decoder's first x).
    const int ts = (t + 1 < Tc) ? (t + 1) : (Tc - 1);
    for (int idx = tid; idx < 16 * Fc; idx += 256) {
      const int r = idx >> 5, c = idx & 31;
      An[r * ROWS + c] =
          f2bf(input[(size_t)(m0 + r) * (Tc * Fc) + (size_t)ts * Fc + c]);
    }
    lstm_wave_step(encW, enc_b, Ac, An, cst, wave, lane);
    __syncthreads();
  }

  // -------- Decoder: 48 steps, projection feeds x back --------
  for (int p = 0; p < Pc; ++p) {
    const unsigned short* Ac = Abuf[p & 1];
    unsigned short*       An = Abuf[(p + 1) & 1];
    lstm_wave_step(decW, dec_b, Ac, An, cst, wave, lane);
    __syncthreads();                 // h complete in An
    if (wave < 2) {
      proj_step(linW, lin_b, An, out, p, m0, wave, lane);
    }
    __syncthreads();                 // feedback x complete in An
  }
}

// ---------------------------------------------------------------------------
// Host launch
// ---------------------------------------------------------------------------
extern "C" void kernel_launch(void* const* d_in, const int* in_sizes, int n_in,
                              void* d_out, int out_size, void* d_ws, size_t ws_size,
                              hipStream_t stream) {
  const float* input   = (const float*)d_in[0];
  // d_in[1] = target (unused), d_in[2] = pred_len (compile-time Pc)
  const float* enc_Wih = (const float*)d_in[3];
  const float* enc_Whh = (const float*)d_in[4];
  const float* enc_b   = (const float*)d_in[5];
  const float* dec_Wih = (const float*)d_in[6];
  const float* dec_Whh = (const float*)d_in[7];
  const float* dec_b   = (const float*)d_in[8];
  const float* lin_W   = (const float*)d_in[9];
  const float* lin_b   = (const float*)d_in[10];

  // Workspace: packed bf16 B-fragments.
  // enc/dec: 32 nt x 4 gates x 17 kt x 1KB = 2.125 MB each; lin: 2 x 16 x 1KB.
  unsigned int* encP = (unsigned int*)d_ws;
  unsigned int* decP = encP + (size_t)128 * NKT * 256;
  unsigned int* linP = decP + (size_t)128 * NKT * 256;

  {
    const int total = 32 * 4 * NKT * 32;       // one thread per fragment-lane
    const int blocks = (total + 255) / 256;
    pack_wmma_b<<<blocks, 256, 0, stream>>>(enc_Wih, enc_Whh, Fc, Hc, 4, 32, NKT, encP);
    pack_wmma_b<<<blocks, 256, 0, stream>>>(dec_Wih, dec_Whh, Fc, Hc, 4, 32, NKT, decP);
  }
  {
    const int total = 2 * 1 * 16 * 32;
    const int blocks = (total + 255) / 256;
    pack_wmma_b<<<blocks, 256, 0, stream>>>(nullptr, lin_W, 0, Hc, 1, 2, 16, linP);
  }

  seq2seq_kernel<<<Bc / 16, 256, 0, stream>>>(
      input, enc_b, dec_b, lin_b,
      (const u32x4*)encP, (const u32x4*)decP, (const u32x4*)linP,
      (float*)d_out);
}
